// NonLocalMeansGray_2156073583167
// MI455X (gfx1250) — matrix operand order repaired and checked by reading, not compiled, not run on hardware
//
#include <hip/hip_runtime.h>

typedef __attribute__((ext_vector_type(16))) _Float16 v16h;
typedef __attribute__((ext_vector_type(8)))  _Float16 v8h;
typedef __attribute__((ext_vector_type(8)))  float    v8f;

namespace {
constexpr int IMG   = 512;
constexpr int TILE  = 32;
constexpr int RAD   = 10;               // search radius
constexpr int PRAD  = 3;                // patch radius
constexpr int HALO  = RAD + PRAD;       // 13
constexpr int YDIM  = TILE + 2 * HALO;  // 58
constexpr int YP    = 60;               // luminance LDS pitch
constexpr int RGDIM = TILE + 2 * RAD;   // 52
constexpr int RGP   = 54;               // rgb LDS pitch
constexpr int SP    = 24;               // scratch pitch in halves (48B rows, 16B aligned)
}

__global__ __launch_bounds__(128)
void nlm_wmma_kernel(const float* __restrict__ rgb,
                     const float* __restrict__ hptr,
                     float* __restrict__ out)
{
    __shared__ float yl[YDIM * YP];
    __shared__ float rgl[3][RGDIM * RGP];
    __shared__ alignas(16) _Float16 scratch[4 * 16 * SP];
    __shared__ alignas(16) _Float16 zeroblk[8];

    const int tid  = threadIdx.x;
    const int lane = tid & 31;
    const int wv   = tid >> 5;
    const int j    = lane & 15;   // column index inside 16x16 sub-tile (and A-row for stage 2)
    const int hx   = lane >> 4;   // lane half

    const int b  = blockIdx.x >> 8;       // 256 tiles per image
    const int t  = blockIdx.x & 255;
    const int ty = (t >> 4) * TILE;
    const int tx = (t & 15) * TILE;

    const float* img = rgb + (size_t)b * 3 * IMG * IMG;

    // ---- stage clipped luminance tile (halo 13, circular wrap) ----
    for (int idx = tid; idx < YDIM * YDIM; idx += 128) {
        const int r = idx / YDIM, c = idx - r * YDIM;
        const int gr = (ty + r - HALO) & (IMG - 1);
        const int gc = (tx + c - HALO) & (IMG - 1);
        const float R = fminf(fmaxf(img[0 * IMG * IMG + gr * IMG + gc], 0.f), 1.f);
        const float G = fminf(fmaxf(img[1 * IMG * IMG + gr * IMG + gc], 0.f), 1.f);
        const float B = fminf(fmaxf(img[2 * IMG * IMG + gr * IMG + gc], 0.f), 1.f);
        yl[r * YP + c] = 0.299f * R + 0.587f * G + 0.114f * B;
    }
    // ---- stage clipped rgb tile (halo 10, circular wrap) ----
    for (int idx = tid; idx < 3 * RGDIM * RGDIM; idx += 128) {
        const int ch = idx / (RGDIM * RGDIM);
        const int rc = idx - ch * RGDIM * RGDIM;
        const int r = rc / RGDIM, c = rc - r * RGDIM;
        const int gr = (ty + r - RAD) & (IMG - 1);
        const int gc = (tx + c - RAD) & (IMG - 1);
        rgl[ch][r * RGP + c] = fminf(fmaxf(img[ch * IMG * IMG + gr * IMG + gc], 0.f), 1.f);
    }
    for (int idx = tid; idx < 4 * 16 * SP; idx += 128) scratch[idx] = (_Float16)0.f;
    if (tid < 8) zeroblk[tid] = (_Float16)0.f;
    __syncthreads();

    // ---- constant band fragments (built once, live in VGPRs) ----
    // bandA K index (contiguous, same order as fa): lanes<16 K=2v,2v+1  ; lanes>=16 K=16+2v,17+2v
    // B-layout K index: lanes<16 v<4:K=2v..  v>=4:K=2v+8 ; lanes>=16 +8
    v16h bandA, bandB;
#pragma unroll
    for (int v = 0; v < 8; ++v) {
#pragma unroll
        for (int p = 0; p < 2; ++p) {
            const int kA = 2 * v + p + 16 * hx;
            const int kB = 2 * v + p + (v >= 4 ? 8 : 0) + 8 * hx;
            bandA[2 * v + p] = (_Float16)((kA >= j && kA <= j + 6) ? 1.f : 0.f);
            bandB[2 * v + p] = (_Float16)((kB >= j && kB <= j + 6) ? 1.f : 0.f);
        }
    }

    // ---- per-wave 16x16 sub-tile ----
    const int r0  = (wv >> 1) * 16;
    const int c0  = (wv & 1) * 16;
    const int rA0 = r0 - PRAD;
    const int cA  = c0 - PRAD + j + HALO;     // frag-a column (halo coords)
    const int cB  = c0 + 13 + j + HALO;       // frag-b column (cols >= 22 unused)
    const int cBu = (cB > YDIM - 1) ? (YDIM - 1) : cB;

    // cache unshifted luminance columns for this lane's B-fragment slots
    float ycA[16], ycB[16];
#pragma unroll
    for (int v = 0; v < 8; ++v) {
#pragma unroll
        for (int p = 0; p < 2; ++p) {
            const int k   = 2 * v + p + 16 * hx;
            const bool ok = (k < 22);
            const int rr  = rA0 + (ok ? k : 21) + HALO;
            ycA[2 * v + p] = ok ? yl[rr * YP + cA]  : 0.f;
            ycB[2 * v + p] = ok ? yl[rr * YP + cBu] : 0.f;
        }
    }

    float den[8], nR[8], nG[8], nB[8];
#pragma unroll
    for (int v = 0; v < 8; ++v) { den[v] = 0.f; nR[v] = 0.f; nG[v] = 0.f; nB[v] = 0.f; }

    const float invh   = 1.f / (fmaxf(hptr[0], 0.f) + 1e-8f);
    const float escale = -1.4426950408889634f * invh;   // exp(x) = exp2(x*log2e)

    _Float16* sc = scratch + wv * 16 * SP;
    const v8h* ap0 = (const v8h*)(sc + j * SP + (hx ? 8 : 0));
    const v8h* ap1 = hx ? (const v8h*)zeroblk : (const v8h*)(sc + j * SP + 16);

#pragma unroll 1
    for (int s = 0; s < 441; ++s) {
        const int dy = s / 21 - RAD;
        const int dx = s - (s / 21) * 21 - RAD;

        // diff^2 B-fragments: lane = column, 16 consecutive rows of (y - y_shift)^2
        v16h fa, fb;
#pragma unroll
        for (int v = 0; v < 8; ++v) {
#pragma unroll
            for (int p = 0; p < 2; ++p) {
                const int k   = 2 * v + p + 16 * hx;
                const bool ok = (k < 22);
                const int rr  = rA0 + (ok ? k : 21) + dy + HALO;
                const int ca  = cA + dx;
                int cb        = cB + dx; cb = (cb > YDIM - 1) ? (YDIM - 1) : cb;
                const float da = ycA[2 * v + p] - yl[rr * YP + ca];
                const float db = ycB[2 * v + p] - yl[rr * YP + cb];
                fa[2 * v + p] = (_Float16)(ok ? da * da : 0.f);
                fb[2 * v + p] = (_Float16)(ok ? db * db : 0.f);
            }
        }

        const v8f cz = {};
        // stage 1: vertical 7-tap box sum = BandA(16x32) x Diff2(32x16)
        v8f d1a = __builtin_amdgcn_wmma_f32_16x16x32_f16(false, bandA, false, fa,
                                                         (short)0, cz, false, false);
        v8f d1b = __builtin_amdgcn_wmma_f32_16x16x32_f16(false, bandA, false, fb,
                                                         (short)0, cz, false, false);

        // D-layout -> A-layout round trip through per-wave LDS scratch (f16)
        const int rb = 8 * hx;
#pragma unroll
        for (int v = 0; v < 8; ++v)
            sc[(rb + v) * SP + j] = (_Float16)d1a[v];
        if (j < 6) {
#pragma unroll
            for (int v = 0; v < 8; ++v)
                sc[(rb + v) * SP + 16 + j] = (_Float16)d1b[v];
        }

        const v8h lo8 = *ap0;
        const v8h hi8 = *ap1;
        v16h a2;
#pragma unroll
        for (int e = 0; e < 8; ++e) { a2[e] = lo8[e]; a2[8 + e] = hi8[e]; }

        // stage 2: horizontal 7-tap box sum = V(16x32) x BandB(32x16)
        v8f d2 = __builtin_amdgcn_wmma_f32_16x16x32_f16(false, a2, false, bandB,
                                                        (short)0, cz, false, false);

        // weights + accumulation (VALU/TRANS, co-executes with matrix pipe)
        const int prow = r0 + 8 * hx + dy + RAD;
        const int pcol = c0 + j + dx + RAD;
#pragma unroll
        for (int v = 0; v < 8; ++v) {
            const float box  = fmaxf(d2[v], 0.f);
            const float dist = __builtin_amdgcn_sqrtf(box);
            const float w    = __builtin_amdgcn_exp2f(dist * escale);
            const int off    = (prow + v) * RGP + pcol;
            den[v] += w;
            nR[v] = __builtin_fmaf(rgl[0][off], w, nR[v]);
            nG[v] = __builtin_fmaf(rgl[1][off], w, nG[v]);
            nB[v] = __builtin_fmaf(rgl[2][off], w, nB[v]);
        }
    }

    // ---- writeback: clip(num/den, 0, 1) ----
    float* op = out + (size_t)b * 3 * IMG * IMG;
    const int gy0 = ty + r0 + 8 * hx;
    const int gx  = tx + c0 + j;
#pragma unroll
    for (int v = 0; v < 8; ++v) {
        const int gy    = gy0 + v;
        const float rcp = 1.f / den[v];
        op[0 * IMG * IMG + gy * IMG + gx] = fminf(fmaxf(nR[v] * rcp, 0.f), 1.f);
        op[1 * IMG * IMG + gy * IMG + gx] = fminf(fmaxf(nG[v] * rcp, 0.f), 1.f);
        op[2 * IMG * IMG + gy * IMG + gx] = fminf(fmaxf(nB[v] * rcp, 0.f), 1.f);
    }
}

extern "C" void kernel_launch(void* const* d_in, const int* in_sizes, int n_in,
                              void* d_out, int out_size, void* d_ws, size_t ws_size,
                              hipStream_t stream)
{
    (void)in_sizes; (void)n_in; (void)out_size; (void)d_ws; (void)ws_size;
    const float* rgb = (const float*)d_in[0];
    const float* h   = (const float*)d_in[1];
    float* out       = (float*)d_out;
    // 8 images x (16x16) tiles of 32x32 = 2048 blocks, 128 threads (4 waves)
    nlm_wmma_kernel<<<dim3(8 * 256), dim3(128), 0, stream>>>(rgb, h, out);
}
